// PropagationGNN_13735305413329
// MI455X (gfx1250) — matrix operand order repaired and checked by reading, not compiled, not run on hardware
//
#include <hip/hip_runtime.h>
#include <hip/hip_bf16.h>

// ---------------------------------------------------------------------------
// 3-layer GAT for MI455X (gfx1250, wave32).
//  - GEMMs via v_wmma_f32_16x16x32_bf16; ping-pong double-buffered LDS tiles
//    filled with global_load_async_to_lds_b128 (ASYNCcnt) overlapped with WMMA
//  - W pre-transposed at bf16-conversion time so A and B WMMA fragments are
//    contiguous 16B runs in LDS (ds_load_b128)
//  - edge softmax: segment-max (uint-keyed atomicMax) -> den pass -> aggregate pass
//  - finalize: bias + LayerNorm + ELU, one 256-thread block per node
// ---------------------------------------------------------------------------

typedef __attribute__((ext_vector_type(16))) __bf16    v16bf;
typedef __attribute__((ext_vector_type(8)))  float     v8f;
typedef __attribute__((ext_vector_type(4)))  unsigned  u4v;
typedef __attribute__((ext_vector_type(4)))  int       i4v;
// async-copy pointer types: clang declares the builtin as
// (i4v addrspace(1)*, i4v addrspace(3)*, imm, imm)
typedef __attribute__((address_space(1))) i4v i4v_glob;
typedef __attribute__((address_space(3))) i4v i4v_lds;

#define HIDF 256

#if __has_builtin(__builtin_amdgcn_global_load_async_to_lds_b128)
#define HAVE_ASYNC_LDS 1
#else
#define HAVE_ASYNC_LDS 0
#endif

__device__ __forceinline__ void wait_asynccnt0() {
#if HAVE_ASYNC_LDS
#if __has_builtin(__builtin_amdgcn_s_wait_asynccnt)
  __builtin_amdgcn_s_wait_asynccnt(0);
#else
  asm volatile("s_wait_asynccnt 0x0" ::: "memory");
#endif
#endif
}

__device__ __forceinline__ unsigned short f32_to_bf16_rne(float f) {
  unsigned u = __float_as_uint(f);
  unsigned r = 0x7FFFu + ((u >> 16) & 1u);
  return (unsigned short)((u + r) >> 16);
}
__device__ __forceinline__ unsigned fOrd(float f) {
  unsigned b = __float_as_uint(f);
  return (b & 0x80000000u) ? ~b : (b | 0x80000000u);
}
__device__ __forceinline__ float ordToF(unsigned k) {
  unsigned b = (k & 0x80000000u) ? (k ^ 0x80000000u) : ~k;
  return __uint_as_float(b);
}
__device__ __forceinline__ float lrelu(float x) { return x > 0.0f ? x : 0.2f * x; }

// ---------------- conversion / zero ----------------------------------------
__global__ void cvt_bf16_kernel(const float* __restrict__ in,
                                unsigned short* __restrict__ out, long long n) {
  long long i = (long long)blockIdx.x * blockDim.x + threadIdx.x;
  long long stride = (long long)gridDim.x * blockDim.x;
  for (; i < n; i += stride) out[i] = f32_to_bf16_rne(in[i]);
}

// W [K x 256] f32 -> Wt [256 x K] bf16 (so B tiles are column-major in K)
__global__ void cvt_bf16_transpose_kernel(const float* __restrict__ in,
                                          unsigned short* __restrict__ out,
                                          int K) {
  int idx = blockIdx.x * blockDim.x + threadIdx.x;
  if (idx >= K * HIDF) return;
  int k = idx >> 8, n = idx & (HIDF - 1);
  out[(long long)n * K + k] = f32_to_bf16_rne(in[idx]);
}

__global__ void zero_kernel(unsigned* __restrict__ p, long long n) {
  long long i = (long long)blockIdx.x * blockDim.x + threadIdx.x;
  long long stride = (long long)gridDim.x * blockDim.x;
  for (; i < n; i += stride) p[i] = 0u;
}

// ---------------- WMMA GEMM: C[M x 256] = A[M x K] * Bt[256 x K]^T ---------
// block tile 64x64, 8 waves (4 M-tiles x 2 N-halves), K step 32 (bf16),
// double-buffered async global->LDS fills overlapped with WMMA.
#define MBLK 64
#define NBLK 64
#define KSTEP 32
__global__ __launch_bounds__(256) void gemm_bf16_wmma_kernel(
    const unsigned short* __restrict__ A,   // [M x K] row-major
    const unsigned short* __restrict__ Bt,  // [256 x K] (W transposed)
    float* __restrict__ C, int M, int K) {
  __shared__ unsigned short As[2][MBLK][KSTEP];   // 2 x 4 KB
  __shared__ unsigned short Bst[2][NBLK][KSTEP];  // 2 x 4 KB
  const int tid = threadIdx.x;
  const int lane = tid & 31;
  const int wave = tid >> 5;   // 0..7
  const int wm = wave >> 1;    // 0..3 : 16-row tile
  const int wn = wave & 1;     // 0..1 : 32-col half
  const int half = lane >> 4;  // 0/1
  const int l15 = lane & 15;
  const int blockM = blockIdx.x * MBLK;
  const int blockN = blockIdx.y * NBLK;

  // tile-fill assignment: 256 threads x 16B = one full 4KB tile each
  const int frow = tid >> 2;        // 0..63 (A row / B column within tile)
  const int fseg = (tid & 3) * 8;   // ushort offset of 16B segment
  int garow = blockM + frow;
  if (garow >= M) garow = M - 1;    // clamp: junk rows never stored
  const unsigned short* agp = A + (long long)garow * K + fseg;
  const unsigned short* bgp = Bt + (long long)(blockN + frow) * K + fseg;

  auto issue_fill = [&](int buf, int k0) {
#if HAVE_ASYNC_LDS
    __builtin_amdgcn_global_load_async_to_lds_b128(
        (i4v_glob*)(agp + k0), (i4v_lds*)&As[buf][frow][fseg], 0, 0);
    __builtin_amdgcn_global_load_async_to_lds_b128(
        (i4v_glob*)(bgp + k0), (i4v_lds*)&Bst[buf][frow][fseg], 0, 0);
#else
    *(u4v*)&As[buf][frow][fseg]  = *(const u4v*)(agp + k0);
    *(u4v*)&Bst[buf][frow][fseg] = *(const u4v*)(bgp + k0);
#endif
  };

  v8f acc0 = {};
  v8f acc1 = {};

  issue_fill(0, 0);  // prologue: tile 0 in flight

  for (int k0 = 0, it = 0; k0 < K; k0 += KSTEP, ++it) {
    const int cur = it & 1;
    // wait for the copy targeting buf[cur] (the only outstanding pair),
    // then make it visible to all waves. The s_wait_dscnt before each WMMA
    // guarantees buf reads are done before the next barrier, so one barrier
    // per iteration is sufficient for the ping-pong.
    wait_asynccnt0();
    __syncthreads();
    if (k0 + KSTEP < K) issue_fill(cur ^ 1, k0 + KSTEP);  // overlap with WMMA

    // A fragment (16x32): lanes 0-15 row=l15 K=0..15,
    // lanes 16-31 row=l15 K=16..31 -> two 16B LDS loads per lane
    union { u4v u[2]; v16bf v; } af;
    {
      const u4v* arp = (const u4v*)(&As[cur][wm * 16 + l15][0]);
      af.u[0] = arp[half * 2];
      af.u[1] = arp[half * 2 + 1];
    }
    // B fragment (32x16): lanes 0-15 K=0..15, lanes 16-31 K=16..31, N=l15
    // column-major tile -> two 16B LDS loads per lane
    union { u4v u[2]; v16bf v; } bf0, bf1;
    {
      const u4v* brp0 = (const u4v*)(&Bst[cur][wn * 32 + l15][0]);
      bf0.u[0] = brp0[half * 2];
      bf0.u[1] = brp0[half * 2 + 1];
      const u4v* brp1 = (const u4v*)(&Bst[cur][wn * 32 + 16 + l15][0]);
      bf1.u[0] = brp1[half * 2];
      bf1.u[1] = brp1[half * 2 + 1];
    }
    acc0 = __builtin_amdgcn_wmma_f32_16x16x32_bf16(false, af.v, false, bf0.v,
                                                   (short)0, acc0, false, false);
    acc1 = __builtin_amdgcn_wmma_f32_16x16x32_bf16(false, af.v, false, bf1.v,
                                                   (short)0, acc1, false, false);
  }

  // C/D layout: VGPR g, lane l -> M = g + 8*(l>>4), N = l&15
  const long long cbase =
      (long long)(blockM + wm * 16 + 8 * half) * HIDF + blockN + wn * 32 + l15;
  if (blockM + MBLK <= M) {
#pragma unroll
    for (int g = 0; g < 8; ++g) {
      C[cbase + (long long)g * HIDF]      = acc0[g];
      C[cbase + (long long)g * HIDF + 16] = acc1[g];
    }
  } else {
#pragma unroll
    for (int g = 0; g < 8; ++g) {
      int gr = blockM + wm * 16 + 8 * half + g;
      if (gr < M) {
        C[cbase + (long long)g * HIDF]      = acc0[g];
        C[cbase + (long long)g * HIDF + 16] = acc1[g];
      }
    }
  }
}

// ---------------- per-node attention dots ----------------------------------
__global__ void alpha_kernel(const float* __restrict__ h,
                             const float* __restrict__ a_src,
                             const float* __restrict__ a_dst,
                             float* __restrict__ asrc, float* __restrict__ adst,
                             int n, int heads, int ch) {
  int idx = blockIdx.x * blockDim.x + threadIdx.x;
  if (idx >= n * heads) return;
  int node = idx / heads, hh = idx - node * heads;
  const float* hp = h + (long long)node * HIDF + hh * ch;
  const float* as = a_src + hh * ch;
  const float* ad = a_dst + hh * ch;
  float s0 = 0.f, s1 = 0.f;
  for (int c = 0; c < ch; ++c) {
    float v = hp[c];
    s0 += v * as[c];
    s1 += v * ad[c];
  }
  asrc[idx] = s0;
  adst[idx] = s1;
}

// ---------------- edge pass 1: segment max ---------------------------------
__global__ void edge_max_kernel(const int* __restrict__ ei, int E, int n, int heads,
                                const float* __restrict__ asrc,
                                const float* __restrict__ adst,
                                unsigned* __restrict__ mkey) {
  long long total = (long long)(E + n) * heads;
  long long w = (long long)blockIdx.x * blockDim.x + threadIdx.x;
  if (w >= total) return;
  long long eidx = w / heads;
  int hh = (int)(w - eidx * heads);
  int s, d;
  if (eidx < E) { s = ei[eidx]; d = ei[(long long)E + eidx]; }
  else          { s = d = (int)(eidx - E); }
  float e = lrelu(asrc[s * heads + hh] + adst[d * heads + hh]);
  atomicMax(&mkey[d * heads + hh], fOrd(e));
}

// ---------------- edge pass 2a: denominator --------------------------------
__global__ void edge_den_kernel(const int* __restrict__ ei, int E, int n, int heads,
                                const float* __restrict__ asrc,
                                const float* __restrict__ adst,
                                const unsigned* __restrict__ mkey,
                                float* __restrict__ den) {
  long long total = (long long)(E + n) * heads;
  long long w = (long long)blockIdx.x * blockDim.x + threadIdx.x;
  if (w >= total) return;
  long long eidx = w / heads;
  int hh = (int)(w - eidx * heads);
  int s, d;
  if (eidx < E) { s = ei[eidx]; d = ei[(long long)E + eidx]; }
  else          { s = d = (int)(eidx - E); }
  float e = lrelu(asrc[s * heads + hh] + adst[d * heads + hh]);
  float ex = __expf(e - ordToF(mkey[d * heads + hh]));
  atomicAdd(&den[d * heads + hh], ex);
}

// ---------------- edge pass 2b: weighted aggregate (numerator) -------------
__global__ void edge_agg_kernel(const int* __restrict__ ei, int E, int n, int heads,
                                int ch, const float* __restrict__ asrc,
                                const float* __restrict__ adst,
                                const unsigned* __restrict__ mkey,
                                const float* __restrict__ h,
                                float* __restrict__ num) {
  long long total = (long long)(E + n) * HIDF;
  long long w = (long long)blockIdx.x * blockDim.x + threadIdx.x;
  if (w >= total) return;
  long long eidx = w >> 8;
  int t = (int)(w & (HIDF - 1));
  int hh = t / ch;
  int s, d;
  if (eidx < E) { s = ei[eidx]; d = ei[(long long)E + eidx]; }
  else          { s = d = (int)(eidx - E); }
  float e = lrelu(asrc[s * heads + hh] + adst[d * heads + hh]);
  float ex = __expf(e - ordToF(mkey[d * heads + hh]));
  atomicAdd(&num[(long long)d * HIDF + t], ex * h[(long long)s * HIDF + t]);
}

// ---------------- finalize: num/den + bias, LayerNorm, ELU -----------------
__global__ __launch_bounds__(256) void finalize_ln_elu_kernel(
    const float* __restrict__ num, const float* __restrict__ den,
    const float* __restrict__ bias, const float* __restrict__ lnw,
    const float* __restrict__ lnb, float* __restrict__ out, int heads, int ch) {
  const int node = blockIdx.x;
  const int t = threadIdx.x;
  const int hh = t / ch;
  float v = num[(long long)node * HIDF + t] / den[node * heads + hh] + bias[t];

  __shared__ float red[256];
  red[t] = v;
  __syncthreads();
  for (int s = 128; s > 0; s >>= 1) {
    if (t < s) red[t] += red[t + s];
    __syncthreads();
  }
  float mu = red[0] * (1.0f / 256.0f);
  __syncthreads();
  float dv = v - mu;
  red[t] = dv * dv;
  __syncthreads();
  for (int s = 128; s > 0; s >>= 1) {
    if (t < s) red[t] += red[t + s];
    __syncthreads();
  }
  float var = red[0] * (1.0f / 256.0f);
  float y = dv * rsqrtf(var + 1e-5f) * lnw[t] + lnb[t];
  out[(long long)node * HIDF + t] = y > 0.0f ? y : (__expf(y) - 1.0f);
}

// ---------------------------------------------------------------------------
extern "C" void kernel_launch(void* const* d_in, const int* in_sizes, int n_in,
                              void* d_out, int out_size, void* d_ws, size_t ws_size,
                              hipStream_t stream) {
  const float* x = (const float*)d_in[0];
  const int* ei = (const int*)d_in[1];
  const float* Wl[3]  = {(const float*)d_in[2],  (const float*)d_in[8],  (const float*)d_in[14]};
  const float* Asl[3] = {(const float*)d_in[3],  (const float*)d_in[9],  (const float*)d_in[15]};
  const float* Adl[3] = {(const float*)d_in[4],  (const float*)d_in[10], (const float*)d_in[16]};
  const float* Bl[3]  = {(const float*)d_in[5],  (const float*)d_in[11], (const float*)d_in[17]};
  const float* Lw[3]  = {(const float*)d_in[6],  (const float*)d_in[12], (const float*)d_in[18]};
  const float* Lb[3]  = {(const float*)d_in[7],  (const float*)d_in[13], (const float*)d_in[19]};

  const int N = in_sizes[0] / 512;
  const int E = in_sizes[1] / 2;
  const int heads[3] = {4, 4, 1};
  const int fin[3] = {512, 256, 256};

  // workspace layout (aliased; ~106 MB total):
  //  R1 (max(N*512*2, N*256*4)): xbf  <->  num
  //  R2 (N*256*4):               h    <->  next-layer input
  //  R3: wbf (512*256*2, transposed)
  //  R4: asrc, adst, mkey, den (N*4 each)
  char* ws = (char*)d_ws;
  size_t R1 = (size_t)N * 512 * 2;
  size_t R2 = (size_t)N * HIDF * 4;
  size_t R3 = (size_t)512 * HIDF * 2;
  unsigned short* xbf = (unsigned short*)ws;
  float* num          = (float*)ws;
  float* hbuf         = (float*)(ws + R1);
  unsigned short* wbf = (unsigned short*)(ws + R1 + R2);
  float* asrc = (float*)(ws + R1 + R2 + R3);
  float* adst = asrc + (size_t)N * 4;
  unsigned* mkey = (unsigned*)(adst + (size_t)N * 4);
  float* den = (float*)(mkey + (size_t)N * 4);

  const float* curx = x;
  for (int L = 0; L < 3; ++L) {
    const int K = fin[L], H = heads[L], C = HIDF / H;
    const long long nEdge = (long long)E + N;

    // convert input to bf16; convert+transpose weights to bf16
    {
      long long n = (long long)N * K;
      cvt_bf16_kernel<<<(int)((n + 255) / 256), 256, 0, stream>>>(curx, xbf, n);
      int nw = K * HIDF;
      cvt_bf16_transpose_kernel<<<(nw + 255) / 256, 256, 0, stream>>>(Wl[L], wbf, K);
    }
    // GEMM h = x @ W  (WMMA bf16 -> f32)
    {
      dim3 grid((N + MBLK - 1) / MBLK, HIDF / NBLK);
      gemm_bf16_wmma_kernel<<<grid, 256, 0, stream>>>(xbf, wbf, hbuf, N, K);
    }
    // attention dots
    alpha_kernel<<<(N * H + 255) / 256, 256, 0, stream>>>(hbuf, Asl[L], Adl[L],
                                                          asrc, adst, N, H, C);
    // zero num (aliases xbf region) + mkey + den (contiguous)
    {
      long long n = (long long)N * HIDF;
      zero_kernel<<<(int)((n + 255) / 256), 256, 0, stream>>>((unsigned*)num, n);
      long long m = (long long)N * 4 * 2;  // mkey + den
      zero_kernel<<<(int)((m + 255) / 256), 256, 0, stream>>>(mkey, m);
    }
    // edge passes
    {
      long long t1 = nEdge * H;
      edge_max_kernel<<<(int)((t1 + 255) / 256), 256, 0, stream>>>(ei, E, N, H,
                                                                   asrc, adst, mkey);
      edge_den_kernel<<<(int)((t1 + 255) / 256), 256, 0, stream>>>(ei, E, N, H,
                                                                   asrc, adst, mkey, den);
      long long t2 = nEdge * HIDF;
      edge_agg_kernel<<<(int)((t2 + 255) / 256), 256, 0, stream>>>(
          ei, E, N, H, C, asrc, adst, mkey, hbuf, num);
    }
    // finalize: bias + LN + ELU -> next input (aliases hbuf) or d_out
    float* outp = (L == 2) ? (float*)d_out : hbuf;
    finalize_ln_elu_kernel<<<N, 256, 0, stream>>>(num, den, Bl[L], Lw[L], Lb[L],
                                                  outp, H, C);
    curx = hbuf;
  }
}
